// SparseGraphConvolution_12232066859195
// MI455X (gfx1250) — matrix operand order repaired and checked by reading, not compiled, not run on hardware
//
#include <hip/hip_runtime.h>

typedef __attribute__((ext_vector_type(16))) __bf16 v16bf;
typedef __attribute__((ext_vector_type(8)))  float  v8f;
typedef __attribute__((ext_vector_type(4)))  float  v4f;

#define N_NODES 100000
#define N_EDGES 1600000
#define IN_F    512
#define OUT_F   128

// ---------------------------------------------------------------------------
// Pre-pass: W [512,128] f32 row-major -> Wt [128,512] bf16 (column-major W),
// so a B fragment (16 consecutive K of one N column) is one 32B load.
// ---------------------------------------------------------------------------
__global__ __launch_bounds__(256) void wt_kernel(const float* __restrict__ w,
                                                 __bf16* __restrict__ wt) {
    int idx = blockIdx.x * 256 + threadIdx.x;   // 0 .. 512*128-1
    int k = idx >> 7;
    int n = idx & 127;
    wt[(size_t)n * IN_F + k] = (__bf16)w[idx];
}

// ---------------------------------------------------------------------------
// CSR row pointers from sorted COO rows: row_start[r] = lower_bound(adj_row, r)
// ---------------------------------------------------------------------------
__global__ __launch_bounds__(256) void rowptr_kernel(const int* __restrict__ adj_row,
                                                     int* __restrict__ row_start) {
    int r = blockIdx.x * 256 + threadIdx.x;
    if (r > N_NODES) return;
    int lo = 0, hi = N_EDGES;
    while (lo < hi) {
        int mid = (lo + hi) >> 1;
        if (adj_row[mid] < r) lo = mid + 1; else hi = mid;
    }
    row_start[r] = lo;
}

// ---------------------------------------------------------------------------
// GEMM: support = x @ W via v_wmma_f32_16x16x32_bf16.
// Block = 256 threads = 8 waves, 64x128 tile. Waves: 4 (M subtiles) x 2 (N halves).
// Each wave: 1 A frag (16x32 bf16), 4 B frags, 4 f32 accumulators per K-step.
// ---------------------------------------------------------------------------
__global__ __launch_bounds__(256) void gemm_wmma(const float* __restrict__ x,
                                                 const __bf16* __restrict__ wt,
                                                 float* __restrict__ support) {
    const int lane = threadIdx.x & 31;
    const int wave = threadIdx.x >> 5;   // 0..7
    const int mw   = wave & 3;           // M subtile within block
    const int nh   = wave >> 2;          // N half (0 or 1)
    const int lmod = lane & 15;
    const int lhi  = lane >> 4;          // 0: lanes 0-15, 1: lanes 16-31

    const int m0 = blockIdx.x * 64 + mw * 16;
    const int n0 = nh * 64;

    // A row for this lane (clamped for the partial last tile)
    int mrow = m0 + lmod;
    if (mrow > N_NODES - 1) mrow = N_NODES - 1;
    const float* __restrict__ xrow = x + (size_t)mrow * IN_F;

    // Per-j B fragment base: column (n0 + j*16 + lmod), K base lhi*16
    const __bf16* __restrict__ b0p = wt + (size_t)(n0 +  0 + lmod) * IN_F + lhi * 16;
    const __bf16* __restrict__ b1p = wt + (size_t)(n0 + 16 + lmod) * IN_F + lhi * 16;
    const __bf16* __restrict__ b2p = wt + (size_t)(n0 + 32 + lmod) * IN_F + lhi * 16;
    const __bf16* __restrict__ b3p = wt + (size_t)(n0 + 48 + lmod) * IN_F + lhi * 16;

    v8f acc0 = {}, acc1 = {}, acc2 = {}, acc3 = {};

    for (int ks = 0; ks < IN_F; ks += 32) {
        // --- A fragment: lane-lo K={ks..ks+15}, lane-hi +16 ---
        const float* ap = xrow + ks + lhi * 16;
        v4f a0 = __builtin_nontemporal_load((const v4f*)(ap +  0));
        v4f a1 = __builtin_nontemporal_load((const v4f*)(ap +  4));
        v4f a2 = __builtin_nontemporal_load((const v4f*)(ap +  8));
        v4f a3 = __builtin_nontemporal_load((const v4f*)(ap + 12));
        v16bf a;
        a[0]=(__bf16)a0[0]; a[1]=(__bf16)a0[1]; a[2]=(__bf16)a0[2]; a[3]=(__bf16)a0[3];
        a[4]=(__bf16)a1[0]; a[5]=(__bf16)a1[1]; a[6]=(__bf16)a1[2]; a[7]=(__bf16)a1[3];
        a[8]=(__bf16)a2[0]; a[9]=(__bf16)a2[1]; a[10]=(__bf16)a2[2]; a[11]=(__bf16)a2[3];
        a[12]=(__bf16)a3[0]; a[13]=(__bf16)a3[1]; a[14]=(__bf16)a3[2]; a[15]=(__bf16)a3[3];

        // --- B fragments: 16 consecutive bf16 (one 32B load) per lane ---
        v16bf b0 = *(const v16bf*)(b0p + ks);
        v16bf b1 = *(const v16bf*)(b1p + ks);
        v16bf b2 = *(const v16bf*)(b2p + ks);
        v16bf b3 = *(const v16bf*)(b3p + ks);

        acc0 = __builtin_amdgcn_wmma_f32_16x16x32_bf16(false, a, false, b0, (short)0, acc0, false, false);
        acc1 = __builtin_amdgcn_wmma_f32_16x16x32_bf16(false, a, false, b1, (short)0, acc1, false, false);
        acc2 = __builtin_amdgcn_wmma_f32_16x16x32_bf16(false, a, false, b2, (short)0, acc2, false, false);
        acc3 = __builtin_amdgcn_wmma_f32_16x16x32_bf16(false, a, false, b3, (short)0, acc3, false, false);
    }

    // C/D layout: VGPR r -> (M = r, N = lane) for lanes 0-15, (M = r+8) for 16-31
    #pragma unroll
    for (int r = 0; r < 8; ++r) {
        const int m = m0 + r + lhi * 8;
        if (m < N_NODES) {
            float* orow = support + (size_t)m * OUT_F + lmod;
            orow[n0 +  0] = acc0[r];
            orow[n0 + 16] = acc1[r];
            orow[n0 + 32] = acc2[r];
            orow[n0 + 48] = acc3[r];
        }
    }
}

// ---------------------------------------------------------------------------
// SpMM: one wave32 per node; lane covers 4 channels (32*4 = 128). Gathers hit
// L2 (support = 51.2 MB << 192 MB). No atomics; fully coalesced 512B accesses.
// ---------------------------------------------------------------------------
__global__ __launch_bounds__(256) void spmm_kernel(const int* __restrict__ adj_col,
                                                   const float* __restrict__ adj_vals,
                                                   const float* __restrict__ support,
                                                   const int* __restrict__ row_start,
                                                   float* __restrict__ out) {
    const int lane = threadIdx.x & 31;
    const int node = blockIdx.x * 8 + (threadIdx.x >> 5);
    if (node >= N_NODES) return;

    const int start = row_start[node];
    const int end   = row_start[node + 1];
    const float* __restrict__ sp = support + lane * 4;

    v4f acc = {};
    for (int e = start; e < end; ++e) {
        const int   col = __builtin_amdgcn_readfirstlane(adj_col[e]);
        const float val = adj_vals[e];
        v4f s = *(const v4f*)(sp + (size_t)col * OUT_F);
        acc += val * s;
    }
    __builtin_nontemporal_store(acc, (v4f*)(out + (size_t)node * OUT_F + lane * 4));
}

// ---------------------------------------------------------------------------
extern "C" void kernel_launch(void* const* d_in, const int* in_sizes, int n_in,
                              void* d_out, int out_size, void* d_ws, size_t ws_size,
                              hipStream_t stream) {
    const float* x        = (const float*)d_in[0];
    const int*   adj_row  = (const int*)  d_in[1];
    const int*   adj_col  = (const int*)  d_in[2];
    const float* adj_vals = (const float*)d_in[3];
    const float* weight   = (const float*)d_in[4];
    float*       out      = (float*)d_out;

    // Workspace layout (all 256B-aligned offsets):
    //   [0, 51200000)            support f32 [100000][128]
    //   [51200000, 51331072)     Wt bf16 [128][512]
    //   [51331072, 51731076)     row_start int [100001]
    char*   ws        = (char*)d_ws;
    float*  support   = (float*)ws;
    __bf16* wt        = (__bf16*)(ws + 51200000);
    int*    row_start = (int*)   (ws + 51200000 + 131072);

    wt_kernel     <<<(IN_F * OUT_F) / 256,      256, 0, stream>>>(weight, wt);
    rowptr_kernel <<<(N_NODES + 1 + 255) / 256, 256, 0, stream>>>(adj_row, row_start);
    gemm_wmma     <<<(N_NODES + 63) / 64,       256, 0, stream>>>(x, wt, support);
    spmm_kernel   <<<(N_NODES + 7) / 8,         256, 0, stream>>>(adj_col, adj_vals, support, row_start, out);
}
